// InteractionNetwork_89824946028980
// MI455X (gfx1250) — compile-verified, matrix operand order checked by reading, not run on hardware
//
#include <hip/hip_runtime.h>
#include <hip/hip_bf16.h>

// ---------------------------------------------------------------------------
// GAT (2-layer) for MI455X / gfx1250.  Dense GEMMs via v_wmma_f32_16x16x32_f16
// (wave32, 16x16 tiles), edge softmax/aggregation via L2-resident f32 atomics.
// ---------------------------------------------------------------------------

typedef _Float16 v16h __attribute__((ext_vector_type(16)));
typedef _Float16 v8h  __attribute__((ext_vector_type(8)));
typedef float    v8f  __attribute__((ext_vector_type(8)));

// ---------------- elementwise helpers ----------------

__global__ void k_cvt_f16(const float* __restrict__ in, _Float16* __restrict__ out, int n) {
    int t = blockIdx.x * blockDim.x + threadIdx.x;
    if (t < n) out[t] = (_Float16)in[t];
}

// out[n*K + k] = (f16) in[k*Nn + n]   (store B transposed so B-fragment loads
// match the A-fragment 16-bit WMMA lane layout)
__global__ void k_transpose_cvt(const float* __restrict__ in, _Float16* __restrict__ out,
                                int K, int Nn) {
    int t = blockIdx.x * blockDim.x + threadIdx.x;
    if (t >= K * Nn) return;
    int n = t / K, k = t - n * K;
    out[(size_t)n * K + k] = (_Float16)in[(size_t)k * Nn + n];
}

// agg + bias, relu, convert to f16
__global__ void k_bias_relu_cvt(const float* __restrict__ agg, const float* __restrict__ bias,
                                _Float16* __restrict__ out, int n, int F) {
    int t = blockIdx.x * blockDim.x + threadIdx.x;
    if (t >= n * F) return;
    float v = agg[t] + bias[t % F];
    out[t] = (_Float16)(v > 0.f ? v : 0.f);
}

// ---------------- WMMA GEMM:  C[M,Nn] = A[M,K] x Bt[Nn,K]^T (+bias) ----------
// A, Bt in f16 (Bt stored row = output column), C in f32.
// One wave computes a 16 x (NT*16) tile; 8 waves per 256-thread block.

template <int NT>
__global__ void k_gemm_wmma(const _Float16* __restrict__ A, const _Float16* __restrict__ Bt,
                            float* __restrict__ C, const float* __restrict__ bias,
                            int Mtiles, int Nn, int K) {
    const int wave = threadIdx.x >> 5;
    const int lane = threadIdx.x & 31;
    const int m_tile = blockIdx.x * 8 + wave;
    if (m_tile >= Mtiles) return;                 // whole wave exits uniformly (EXEC all-1 for WMMA)
    const int nbase = blockIdx.y * (NT * 16);

    const int r      = lane & 15;                 // row within 16-row fragment
    const int colsel = (lane >> 4) * 8;           // half-wave K-interleave (16-bit A layout)

    v8f acc[NT];
#pragma unroll
    for (int t = 0; t < NT; ++t) acc[t] = (v8f){};

    const _Float16* arow = A + ((size_t)(m_tile * 16 + r)) * K + colsel;

    for (int k0 = 0; k0 < K; k0 += 32) {
        union { v16h v; v8h h[2]; } a;
        a.h[0] = *(const v8h*)(arow + k0);
        a.h[1] = *(const v8h*)(arow + k0 + 16);
        if (k0 + 32 < K) __builtin_prefetch(arow + k0 + 32, 0, 0);

#pragma unroll
        for (int t = 0; t < NT; ++t) {
            const _Float16* brow = Bt + ((size_t)(nbase + t * 16 + r)) * K + colsel + k0;
            union { v16h v; v8h h[2]; } b;
            b.h[0] = *(const v8h*)(brow);
            b.h[1] = *(const v8h*)(brow + 16);
            acc[t] = __builtin_amdgcn_wmma_f32_16x16x32_f16(
                false, a.v, false, b.v, (short)0, acc[t], false, false);
        }
    }

    // C/D layout: vgpr v, lanes 0-15 -> (M=v, N=lane); lanes 16-31 -> (M=v+8, N=lane-16)
    const int rowoff = (lane >> 4) << 3;
#pragma unroll
    for (int t = 0; t < NT; ++t) {
        const int col = nbase + t * 16 + (lane & 15);
        const float badd = bias ? bias[col] : 0.f;
#pragma unroll
        for (int v = 0; v < 8; ++v) {
            const int row = m_tile * 16 + v + rowoff;
            C[(size_t)row * Nn + col] = acc[t][v] + badd;
        }
    }
}

// ---------------- attention logits:  ls/ld[n,h] = <xl[n,h,:], a_src/dst[h,:]> ----

template <int H, int C>
__global__ void k_logits(const float* __restrict__ xl, const float* __restrict__ a_src,
                         const float* __restrict__ a_dst, float* __restrict__ ls,
                         float* __restrict__ ld, int n) {
    int t = blockIdx.x * blockDim.x + threadIdx.x;
    if (t >= n * H) return;
    const int h    = t & (H - 1);
    const int node = t >> (H == 4 ? 2 : 0);
    const float* xr = xl + (size_t)node * (H * C) + h * C;
    float s = 0.f, d = 0.f;
#pragma unroll 8
    for (int c = 0; c < C; ++c) {
        s = fmaf(xr[c], a_src[h * C + c], s);
        d = fmaf(xr[c], a_dst[h * C + c], d);
    }
    ls[t] = s;
    ld[t] = d;
}

// ---------------- edge kernels ----------------

__device__ __forceinline__ unsigned enc_f32(float f) {     // order-preserving encode
    unsigned u = __float_as_uint(f);
    return (u & 0x80000000u) ? ~u : (u | 0x80000000u);
}
__device__ __forceinline__ float dec_f32(unsigned e) {
    return __uint_as_float((e & 0x80000000u) ? (e & 0x7FFFFFFFu) : ~e);
}
__device__ __forceinline__ float lrelu(float v) { return v > 0.f ? v : 0.2f * v; }

template <int H>
__global__ void k_edge_max(const int* __restrict__ srcA, const int* __restrict__ dstA,
                           const float* __restrict__ ls, const float* __restrict__ ld,
                           unsigned* __restrict__ m_enc, int E, int N) {
    int t = blockIdx.x * blockDim.x + threadIdx.x;
    if (t >= (E + N) * H) return;
    const int eid = t / H, h = t - eid * H;
    const int src = (eid < E) ? srcA[eid] : (eid - E);
    const int dst = (eid < E) ? dstA[eid] : (eid - E);
    const float e = lrelu(ls[src * H + h] + ld[dst * H + h]);
    atomicMax(&m_enc[dst * H + h], enc_f32(e));
}

template <int H>
__global__ void k_edge_w(const int* __restrict__ srcA, const int* __restrict__ dstA,
                         const float* __restrict__ ls, const float* __restrict__ ld,
                         const unsigned* __restrict__ m_enc, float* __restrict__ denom,
                         float* __restrict__ wbuf, int E, int N) {
    int t = blockIdx.x * blockDim.x + threadIdx.x;
    if (t >= (E + N) * H) return;
    const int eid = t / H, h = t - eid * H;
    const int src = (eid < E) ? srcA[eid] : (eid - E);
    const int dst = (eid < E) ? dstA[eid] : (eid - E);
    const float e = lrelu(ls[src * H + h] + ld[dst * H + h]);
    const float w = __expf(e - dec_f32(m_enc[dst * H + h]));
    wbuf[t] = w;
    atomicAdd(&denom[dst * H + h], w);
}

template <int H, int C>
__global__ void k_edge_agg(const int* __restrict__ srcA, const int* __restrict__ dstA,
                           const float* __restrict__ wbuf, const float* __restrict__ denom,
                           const float* __restrict__ xl, float* __restrict__ agg,
                           int E, int N) {
    const unsigned t = blockIdx.x * blockDim.x + threadIdx.x;
    const unsigned HC = H * C;
    if (t >= (unsigned)(E + N) * HC) return;
    const unsigned eid = t / HC, j = t - eid * HC;
    const unsigned h = j / C;
    const int src = (eid < (unsigned)E) ? srcA[eid] : (int)(eid - E);
    const int dst = (eid < (unsigned)E) ? dstA[eid] : (int)(eid - E);
    const float alpha = wbuf[eid * HC / C + h] / denom[dst * H + h];
    atomicAdd(&agg[(size_t)dst * HC + j], xl[(size_t)src * HC + j] * alpha);
}

// ---------------------------------------------------------------------------

static inline int cdiv(long long a, long long b) { return (int)((a + b - 1) / b); }

extern "C" void kernel_launch(void* const* d_in, const int* in_sizes, int n_in,
                              void* d_out, int out_size, void* d_ws, size_t ws_size,
                              hipStream_t stream) {
    const int D = 128, F1 = 256, F2 = 32, H1 = 4, C1 = 64, H2 = 1, C2 = 32;
    const int N = in_sizes[0] / D;             // 50000 (== 3125 * 16, WMMA-exact)
    const int E = in_sizes[1] / 2;             // 800000
    const int Ep = E + N;                      // + self loops

    const float* x        = (const float*)d_in[0];
    const int*   srcA     = (const int*)d_in[1];
    const int*   dstA     = srcA + E;
    const float* W1       = (const float*)d_in[2];
    const float* att_src1 = (const float*)d_in[3];
    const float* att_dst1 = (const float*)d_in[4];
    const float* b1       = (const float*)d_in[5];
    const float* W2       = (const float*)d_in[6];
    const float* att_src2 = (const float*)d_in[7];
    const float* att_dst2 = (const float*)d_in[8];
    const float* b2       = (const float*)d_in[9];
    const float* fc_W     = (const float*)d_in[10];
    const float* fc_b     = (const float*)d_in[11];
    float* out = (float*)d_out;

    // ---- workspace carve-up (256B aligned) ----
    char* ws = (char*)d_ws;
    size_t off = 0;
    auto alloc = [&](size_t bytes) { char* p = ws + off; off += (bytes + 255) & ~(size_t)255; return p; };
    _Float16* xh    = (_Float16*)alloc((size_t)N * D * 2);
    _Float16* W1t   = (_Float16*)alloc((size_t)F1 * D * 2);
    _Float16* W2t   = (_Float16*)alloc((size_t)F2 * F1 * 2);
    _Float16* fcWt  = (_Float16*)alloc((size_t)D * F2 * 2);
    float*    xl1   = (float*)alloc((size_t)N * F1 * 4);
    float*    ls1   = (float*)alloc((size_t)N * H1 * 4);
    float*    ld1   = (float*)alloc((size_t)N * H1 * 4);
    unsigned* m1    = (unsigned*)alloc((size_t)N * H1 * 4);
    float*    den1  = (float*)alloc((size_t)N * H1 * 4);
    float*    w1    = (float*)alloc((size_t)Ep * H1 * 4);
    float*    agg1  = (float*)alloc((size_t)N * F1 * 4);
    _Float16* h1h   = (_Float16*)alloc((size_t)N * F1 * 2);
    float*    xl2   = (float*)alloc((size_t)N * F2 * 4);
    float*    ls2   = (float*)alloc((size_t)N * H2 * 4);
    float*    ld2   = (float*)alloc((size_t)N * H2 * 4);
    unsigned* m2    = (unsigned*)alloc((size_t)N * H2 * 4);
    float*    den2  = (float*)alloc((size_t)N * H2 * 4);
    float*    w2    = (float*)alloc((size_t)Ep * H2 * 4);
    float*    agg2  = (float*)alloc((size_t)N * F2 * 4);
    _Float16* h2h   = (_Float16*)alloc((size_t)N * F2 * 2);
    (void)ws_size;

    const int TB = 256;
    const int Mtiles = N / 16;                  // 3125
    const int gemmBlocks = cdiv(Mtiles, 8);     // 8 waves per block

    // ---- stage 0: precision conversion / weight transposes ----
    k_cvt_f16<<<cdiv((long long)N * D, TB), TB, 0, stream>>>(x, xh, N * D);
    k_transpose_cvt<<<cdiv(D * F1, TB), TB, 0, stream>>>(W1, W1t, D, F1);
    k_transpose_cvt<<<cdiv(F1 * F2, TB), TB, 0, stream>>>(W2, W2t, F1, F2);
    k_transpose_cvt<<<cdiv(F2 * D, TB), TB, 0, stream>>>(fc_W, fcWt, F2, D);

    // ---- layer 1 ----
    k_gemm_wmma<4><<<dim3(gemmBlocks, F1 / 64), TB, 0, stream>>>(xh, W1t, xl1, nullptr, Mtiles, F1, D);
    k_logits<H1, C1><<<cdiv((long long)N * H1, TB), TB, 0, stream>>>(xl1, att_src1, att_dst1, ls1, ld1, N);
    hipMemsetAsync(m1, 0, (size_t)N * H1 * 4, stream);
    hipMemsetAsync(den1, 0, (size_t)N * H1 * 4, stream);
    hipMemsetAsync(agg1, 0, (size_t)N * F1 * 4, stream);
    k_edge_max<H1><<<cdiv((long long)Ep * H1, TB), TB, 0, stream>>>(srcA, dstA, ls1, ld1, m1, E, N);
    k_edge_w<H1><<<cdiv((long long)Ep * H1, TB), TB, 0, stream>>>(srcA, dstA, ls1, ld1, m1, den1, w1, E, N);
    k_edge_agg<H1, C1><<<cdiv((long long)Ep * H1 * C1, TB), TB, 0, stream>>>(srcA, dstA, w1, den1, xl1, agg1, E, N);
    k_bias_relu_cvt<<<cdiv((long long)N * F1, TB), TB, 0, stream>>>(agg1, b1, h1h, N, F1);

    // ---- layer 2 ----
    k_gemm_wmma<2><<<dim3(gemmBlocks, 1), TB, 0, stream>>>(h1h, W2t, xl2, nullptr, Mtiles, F2, F1);
    k_logits<H2, C2><<<cdiv((long long)N * H2, TB), TB, 0, stream>>>(xl2, att_src2, att_dst2, ls2, ld2, N);
    hipMemsetAsync(m2, 0, (size_t)N * H2 * 4, stream);
    hipMemsetAsync(den2, 0, (size_t)N * H2 * 4, stream);
    hipMemsetAsync(agg2, 0, (size_t)N * F2 * 4, stream);
    k_edge_max<H2><<<cdiv((long long)Ep * H2, TB), TB, 0, stream>>>(srcA, dstA, ls2, ld2, m2, E, N);
    k_edge_w<H2><<<cdiv((long long)Ep * H2, TB), TB, 0, stream>>>(srcA, dstA, ls2, ld2, m2, den2, w2, E, N);
    k_edge_agg<H2, C2><<<cdiv((long long)Ep * H2 * C2, TB), TB, 0, stream>>>(srcA, dstA, w2, den2, xl2, agg2, E, N);
    k_bias_relu_cvt<<<cdiv((long long)N * F2, TB), TB, 0, stream>>>(agg2, b2, h2h, N, F2);

    // ---- final fc (bias fused), writes d_out [N,128] f32 ----
    k_gemm_wmma<4><<<dim3(gemmBlocks, D / 64), TB, 0, stream>>>(h2h, fcWt, out, fc_b, Mtiles, D, F2);
}
